// GraphFourierLayer_6021544149159
// MI455X (gfx1250) — matrix operand order repaired and checked by reading, not compiled, not run on hardware
//
#include <hip/hip_runtime.h>

#define B_ 32
#define N_ 4096
#define C_ 128
#define M_ 256

typedef _Float16 v16h __attribute__((ext_vector_type(16)));
typedef _Float16 v8h  __attribute__((ext_vector_type(8)));
typedef float    v8f  __attribute__((ext_vector_type(8)));

// Build the WMMA f16 A-fragment (contiguous k order, matching B):
// lane (r,h): halves 0..15 = K[kb+16h .. kb+16h+15]
__device__ static inline v16h make_a_frag(const _Float16* __restrict__ row, int kb, int h) {
    return *(const v16h*)(row + kb + 16 * h);
}

__device__ static inline v8f wmma_f16(v16h a, v16h b, v8f c) {
    return __builtin_amdgcn_wmma_f32_16x16x32_f16(false, a, false, b, (short)0, c,
                                                  false, false);
}

// ---- Prep: U [n,m] -> Uh [n,m] f16 (copy) + Ut [m,n] f16 (LDS-tiled transpose) ----
__global__ void conv_u(const float* __restrict__ U, _Float16* __restrict__ Uh,
                       _Float16* __restrict__ Ut) {
    __shared__ float tile[32][33];
    int n0 = blockIdx.x * 32;
    int m0 = blockIdx.y * 32;
#pragma unroll
    for (int j = 0; j < 32; j += 8) {
        float v = U[(size_t)(n0 + threadIdx.y + j) * M_ + m0 + threadIdx.x];
        tile[threadIdx.y + j][threadIdx.x] = v;
        Uh[(size_t)(n0 + threadIdx.y + j) * M_ + m0 + threadIdx.x] = (_Float16)v;
    }
    __syncthreads();
#pragma unroll
    for (int j = 0; j < 32; j += 8)
        Ut[(size_t)(m0 + threadIdx.y + j) * N_ + n0 + threadIdx.x] =
            (_Float16)tile[threadIdx.x][threadIdx.y + j];
}

// ---- Prep: W_real [c,o,m] -> Wt [m,o,c] f16, LDS-tiled per-o transpose ----
__global__ void conv_w(const float* __restrict__ W, _Float16* __restrict__ Wt) {
    __shared__ float tile[32][33];
    int o  = blockIdx.z;       // 0..127
    int m0 = blockIdx.x * 32;  // grid.x = 8
    int c0 = blockIdx.y * 32;  // grid.y = 4
#pragma unroll
    for (int j = 0; j < 32; j += 8)  // read coalesced along m
        tile[threadIdx.y + j][threadIdx.x] =
            W[(size_t)(c0 + threadIdx.y + j) * (C_ * M_) + (size_t)o * M_ + m0 + threadIdx.x];
    __syncthreads();
#pragma unroll
    for (int j = 0; j < 32; j += 8)  // write coalesced along c
        Wt[((size_t)(m0 + threadIdx.y + j) * C_ + o) * C_ + c0 + threadIdx.x] =
            (_Float16)tile[threadIdx.x][threadIdx.y + j];
}

// ---- Prep: x [b,n,c] f32 -> xT [b,c,n] f16 via LDS-tiled transpose ----
__global__ void conv_x(const float* __restrict__ x, _Float16* __restrict__ xT) {
    __shared__ float tile[32][33];
    int b  = blockIdx.z;
    int n0 = blockIdx.x * 32;
    int c0 = blockIdx.y * 32;
    const float* xb = x + (size_t)b * N_ * C_;
#pragma unroll
    for (int j = 0; j < 32; j += 8)
        tile[threadIdx.y + j][threadIdx.x] =
            xb[(size_t)(n0 + threadIdx.y + j) * C_ + c0 + threadIdx.x];
    __syncthreads();
    _Float16* xtb = xT + (size_t)b * C_ * N_;
#pragma unroll
    for (int j = 0; j < 32; j += 8)
        xtb[(size_t)(c0 + threadIdx.y + j) * N_ + n0 + threadIdx.x] =
            (_Float16)tile[threadIdx.x][threadIdx.y + j];
}

// ---- Step 1: xspec[b,m,c] = sum_n Ut[m,n] * xT[b,c,n]   (K = 4096) ----
// One wave per (b, m-tile, c-quarter): 16x32 strip, 2 accumulators. Ping-pong
// double buffer pinned with sched_barrier(0); ~80 live VGPRs fits the 128-VGPR
// allocation with headroom (no spills). 2048 waves.
__global__ void gemm_fwd(const _Float16* __restrict__ Ut, const _Float16* __restrict__ xT,
                         _Float16* __restrict__ xs) {
    int wave = blockIdx.x * (blockDim.x >> 5) + (threadIdx.x >> 5);
    int lane = threadIdx.x & 31;
    int b   = wave >> 6;         // 0..31
    int mt  = (wave >> 2) & 15;  // 0..15
    int cq  = wave & 3;          // c quarter: columns cq*32 .. cq*32+31
    int r = lane & 15, h = lane >> 4;
    const _Float16* arow = Ut + (size_t)(mt * 16 + r) * N_;
    const _Float16* xb   = xT + (size_t)b * C_ * N_ + (size_t)(cq * 32 + r) * N_;
    v8f acc[2] = {};

    v16h a0 = make_a_frag(arow, 0, h);
    v16h b0[2];
#pragma unroll
    for (int ct = 0; ct < 2; ++ct)
        b0[ct] = *(const v16h*)(xb + (size_t)(ct * 16) * N_ + 16 * h);

#pragma unroll 1
    for (int kb = 0; kb < N_; kb += 64) {
        // phase A: issue loads for k+32 into buffer 1
        v16h a1 = make_a_frag(arow, kb + 32, h);
        v16h b1[2];
#pragma unroll
        for (int ct = 0; ct < 2; ++ct)
            b1[ct] = *(const v16h*)(xb + (size_t)(ct * 16) * N_ + kb + 32 + 16 * h);
        __builtin_amdgcn_sched_barrier(0);
        // phase B: compute step k with buffer 0
#pragma unroll
        for (int ct = 0; ct < 2; ++ct) acc[ct] = wmma_f16(a0, b0[ct], acc[ct]);
        __builtin_amdgcn_sched_barrier(0);
        // phase C: issue loads for k+64 (wrapped in-bounds) into buffer 0
        int kn = (kb + 64) & (N_ - 1);
        a0 = make_a_frag(arow, kn, h);
#pragma unroll
        for (int ct = 0; ct < 2; ++ct)
            b0[ct] = *(const v16h*)(xb + (size_t)(ct * 16) * N_ + kn + 16 * h);
        __builtin_amdgcn_sched_barrier(0);
        // phase D: compute step k+32 with buffer 1
#pragma unroll
        for (int ct = 0; ct < 2; ++ct) acc[ct] = wmma_f16(a1, b1[ct], acc[ct]);
    }
#pragma unroll
    for (int ct = 0; ct < 2; ++ct)
#pragma unroll
        for (int i = 0; i < 8; ++i)
            xs[((size_t)b * M_ + mt * 16 + 8 * h + i) * C_ + cq * 32 + ct * 16 + r] =
                (_Float16)acc[ct][i];
}

// ---- Step 2: s2t[b,o,m] = sum_c xs[b,m,c] * Wt[m,o,c]   (K = 128, per mode m) ----
// One wave per (m, batch-half, o-quarter): 16x32 strip, same pinned ping-pong.
__global__ void gemm_mix(const _Float16* __restrict__ xs, const _Float16* __restrict__ Wt,
                         _Float16* __restrict__ s2t) {
    int wave = blockIdx.x * (blockDim.x >> 5) + (threadIdx.x >> 5);
    int lane = threadIdx.x & 31;
    int m  = wave >> 3;         // 0..255
    int bt = (wave >> 2) & 1;   // batch half
    int oq = wave & 3;          // o quarter
    int r = lane & 15, h = lane >> 4;
    const _Float16* arow = xs + ((size_t)(bt * 16 + r) * M_ + m) * C_;
    const _Float16* wm   = Wt + (size_t)m * C_ * C_ + (size_t)(oq * 32 + r) * C_;
    v8f acc[2] = {};

    v16h a0 = make_a_frag(arow, 0, h);
    v16h b0[2];
#pragma unroll
    for (int ot = 0; ot < 2; ++ot)
        b0[ot] = *(const v16h*)(wm + (size_t)(ot * 16) * C_ + 16 * h);

#pragma unroll 1
    for (int kb = 0; kb < C_; kb += 64) {
        v16h a1 = make_a_frag(arow, kb + 32, h);
        v16h b1[2];
#pragma unroll
        for (int ot = 0; ot < 2; ++ot)
            b1[ot] = *(const v16h*)(wm + (size_t)(ot * 16) * C_ + kb + 32 + 16 * h);
        __builtin_amdgcn_sched_barrier(0);
#pragma unroll
        for (int ot = 0; ot < 2; ++ot) acc[ot] = wmma_f16(a0, b0[ot], acc[ot]);
        __builtin_amdgcn_sched_barrier(0);
        int kn = (kb + 64) & (C_ - 1);
        a0 = make_a_frag(arow, kn, h);
#pragma unroll
        for (int ot = 0; ot < 2; ++ot)
            b0[ot] = *(const v16h*)(wm + (size_t)(ot * 16) * C_ + kn + 16 * h);
        __builtin_amdgcn_sched_barrier(0);
#pragma unroll
        for (int ot = 0; ot < 2; ++ot) acc[ot] = wmma_f16(a1, b1[ot], acc[ot]);
    }
#pragma unroll
    for (int ot = 0; ot < 2; ++ot)
#pragma unroll
        for (int i = 0; i < 8; ++i)
            s2t[((size_t)(bt * 16 + 8 * h + i) * C_ + oq * 32 + ot * 16 + r) * M_ + m] =
                (_Float16)acc[ot][i];
}

// ---- Step 3: out[b,n,o] = sum_m Uh[n,m] * s2t[b,o,m]   (K = 256) ----
// One wave per (b, n-tile): 16x128 strip; 8192 waves hide latency by occupancy.
__global__ void gemm_inv(const _Float16* __restrict__ Uh, const _Float16* __restrict__ s2t,
                         float* __restrict__ out) {
    int wave = blockIdx.x * (blockDim.x >> 5) + (threadIdx.x >> 5);
    int lane = threadIdx.x & 31;
    int b  = wave >> 8;   // 0..31
    int nt = wave & 255;  // 0..255
    int r = lane & 15, h = lane >> 4;
    const _Float16* arow = Uh + (size_t)(nt * 16 + r) * M_;
    const _Float16* sb   = s2t + (size_t)b * C_ * M_;
    v8f acc[8] = {};
#pragma unroll
    for (int kb = 0; kb < M_; kb += 32) {
        v16h a = make_a_frag(arow, kb, h);
#pragma unroll
        for (int ot = 0; ot < 8; ++ot) {
            v16h bf = *(const v16h*)(sb + (size_t)(ot * 16 + r) * M_ + kb + 16 * h);
            acc[ot] = wmma_f16(a, bf, acc[ot]);
        }
    }
    float* ob = out + (size_t)b * N_ * C_;
#pragma unroll
    for (int ot = 0; ot < 8; ++ot)
#pragma unroll
        for (int i = 0; i < 8; ++i)
            ob[(size_t)(nt * 16 + 8 * h + i) * C_ + ot * 16 + r] = acc[ot][i];
}

extern "C" void kernel_launch(void* const* d_in, const int* in_sizes, int n_in,
                              void* d_out, int out_size, void* d_ws, size_t ws_size,
                              hipStream_t stream) {
    (void)in_sizes; (void)n_in; (void)out_size; (void)ws_size;
    const float* x  = (const float*)d_in[0];  // [B, N, C_in]
    const float* U  = (const float*)d_in[1];  // [N, M]
    const float* Wr = (const float*)d_in[2];  // [C_in, C_out, M]
    // d_in[3] (W_imag) provably does not affect the output.

    char* ws = (char*)d_ws;
    _Float16* Uh  = (_Float16*)(ws);                        //  2 MB  [N,M]
    _Float16* Ut  = (_Float16*)(ws + ( 2ull << 20));        //  2 MB  [M,N]
    _Float16* Wt  = (_Float16*)(ws + ( 4ull << 20));        //  8 MB  [M,O,C]
    _Float16* xT  = (_Float16*)(ws + (12ull << 20));        // 32 MB  [B,C,N]
    _Float16* xs  = (_Float16*)(ws + (44ull << 20));        //  2 MB  [B,M,C]
    _Float16* s2t = (_Float16*)(ws + (46ull << 20));        //  2 MB  [B,O,M]
    float* out = (float*)d_out;

    conv_u<<<dim3(N_ / 32, M_ / 32), dim3(32, 8), 0, stream>>>(U, Uh, Ut);
    conv_w<<<dim3(M_ / 32, C_ / 32, C_), dim3(32, 8), 0, stream>>>(Wr, Wt);
    conv_x<<<dim3(N_ / 32, C_ / 32, B_), dim3(32, 8), 0, stream>>>(x, xT);

    gemm_fwd<<<256,  256, 0, stream>>>(Ut, xT, xs);    // 2048 waves, K=4096
    gemm_mix<<<256,  256, 0, stream>>>(xs, Wt, s2t);   // 2048 waves, K=128 x 256 modes
    gemm_inv<<<1024, 256, 0, stream>>>(Uh, s2t, out);  // 8192 waves, K=256
}
